// ff_SHD_14989435863736
// MI455X (gfx1250) — compile-verified, matrix operand order checked by reading, not run on hardware
//
#include <hip/hip_runtime.h>

// ---------------------------------------------------------------------------
// SNN forward: Linear(700->512) -> LIF scan -> Linear(512->512) -> LIF scan
//              -> Linear(512->20) summed over T.
// B=128, T=500, D=700 (pad 704), H=512, O=20.  M = T*B = 64000.
// Big GEMMs: v_wmma_f32_16x16x32_bf16, fully unrolled K pipeline (templated K)
// so the 2-deep register double-buffer renames cleanly (no v_mov rotation).
// ---------------------------------------------------------------------------

typedef __attribute__((ext_vector_type(16))) __bf16      v16bf;
typedef __attribute__((ext_vector_type(8)))  float        v8f;
typedef __attribute__((ext_vector_type(4)))  unsigned int v4u;

union FragU { v16bf v; v4u u[2]; };

static constexpr int kB     = 128;
static constexpr int kT     = 500;
static constexpr int kD     = 700;
static constexpr int kDP    = 704;   // D padded to multiple of 32 (and 64)
static constexpr int kH     = 512;
static constexpr int kO     = 20;
static constexpr int kM     = kT * kB;   // 64000 GEMM rows (m = t*128 + b)
static constexpr int kN16   = kH / 16;   // 32 N-tiles of 16

__device__ __forceinline__ unsigned short f2bf(float f) {
  unsigned int u = __builtin_bit_cast(unsigned int, f);
  unsigned int r = (u + 0x7FFFu + ((u >> 16) & 1u)) >> 16;  // RNE
  return (unsigned short)r;
}

// --- pass 0a: x [B,T,D] f32 -> xb [M=t*128+b][704] bf16 (zero-padded K) ----
__global__ __launch_bounds__(256) void k_convert_x(const float* __restrict__ x,
                                                   unsigned short* __restrict__ xb) {
  unsigned int idx = blockIdx.x * 256u + threadIdx.x;
  if (idx >= (unsigned int)kM * kDP) return;
  unsigned int d = idx % kDP;
  unsigned int m = idx / kDP;
  unsigned int t = m >> 7, b = m & 127u;
  float v = 0.f;
  if (d < (unsigned int)kD) v = x[((size_t)b * kT + t) * kD + d];
  xb[idx] = f2bf(v);
}

// --- pass 0b: W [N][Kact] f32 row-major -> WMMA B-fragment blobs -----------
// frag layout: frag[(kc32*kN16 + n16)*512 + lane*16 + i], bf16, where
//   n = n16*16 + (lane&15),  k = kc32*32 + (i<8 ? i : i+8) + (lane>=16 ? 8:0)
__global__ __launch_bounds__(256) void k_wfrag(const float* __restrict__ W,
                                               unsigned short* __restrict__ frag,
                                               int nKc, int Kact, int ldw) {
  int idx = blockIdx.x * 256 + threadIdx.x;
  int total = nKc * kN16 * 512;
  if (idx >= total) return;
  int i    = idx & 15;
  int lane = (idx >> 4) & 31;
  int f    = idx >> 9;
  int n16  = f % kN16;
  int kc   = f / kN16;
  int k = kc * 32 + (i < 8 ? i : i + 8) + ((lane >= 16) ? 8 : 0);
  int n = n16 * 16 + (lane & 15);
  float v = (k < Kact) ? W[(size_t)n * ldw + k] : 0.f;
  frag[idx] = f2bf(v);
}

// --- GEMM helpers ----------------------------------------------------------
__device__ __forceinline__ void loadA(FragU& a, const unsigned short* Arow, int kc) {
  a.u[0] = *(const v4u*)(Arow + kc);        // K = kc+{0..7}    (+8 if laneHi)
  a.u[1] = *(const v4u*)(Arow + kc + 16);   // K = kc+16+{0..7} (+8 if laneHi)
}

__device__ __forceinline__ void loadB4(FragU* b, const unsigned short* bp) {
#pragma unroll
  for (int j = 0; j < 4; ++j) {
    b[j].u[0] = *(const v4u*)(bp + j * 512);
    b[j].u[1] = *(const v4u*)(bp + j * 512 + 8);
  }
}

__device__ __forceinline__ void wmma4(v8f* c, const FragU& a, const FragU* b) {
#pragma unroll
  for (int j = 0; j < 4; ++j) {
    c[j] = __builtin_amdgcn_wmma_f32_16x16x32_bf16(
        false, a.v, false, b[j].v, (short)0, c[j], false, false);
  }
}

// --- bf16 WMMA GEMM: C[M][512] f32 = A[M][KPAD] bf16 * Bfrag + bias --------
// Block = 256 thr = 8 waves; block tile 128(M) x 64(N); wave tile 16 x 64.
// KPAD compile-time (704 / 512) -> K loop fully unrolled, pipeline buffers
// rename per chunk (no register rotation copies).
template <int KPAD>
__global__ __launch_bounds__(256) void k_gemm_bf16(const unsigned short* __restrict__ A,
                                                   const unsigned short* __restrict__ Bf,
                                                   const float* __restrict__ bias,
                                                   float* __restrict__ C) {
  constexpr int NC = KPAD / 32;           // # of 32-wide K chunks
  const int lane   = threadIdx.x & 31;
  const int wave   = threadIdx.x >> 5;
  const int mbase  = blockIdx.x * 128;
  const int nbase  = blockIdx.y * 64;
  const int laneLo = lane & 15;
  const int laneHi = lane >> 4;

  // A: row per lane (16-bit A layout), two 16B chunks per 32-wide K step
  const unsigned short* Arow =
      A + (size_t)(mbase + wave * 16 + laneLo) * KPAD + laneHi * 8;
  // B: per-lane base of this block's N-column fragments; +kStep per K chunk
  constexpr size_t kStep = (size_t)kN16 * 512;
  const unsigned short* Bcol = Bf + (size_t)(nbase >> 4) * 512 + lane * 16;

  v8f c[4] = {v8f{}, v8f{}, v8f{}, v8f{}};

  // two-deep register pipeline, fully unrolled (parity buffers renamed)
  FragU a[2], b[2][4];
  loadA(a[0], Arow, 0);
  loadB4(b[0], Bcol);

#pragma unroll
  for (int ic = 0; ic < NC; ++ic) {
    const int pcur = ic & 1;
    const int pnxt = pcur ^ 1;
    if (ic + 1 < NC) {
      loadA(a[pnxt], Arow, (ic + 1) * 32);
      loadB4(b[pnxt], Bcol + (size_t)(ic + 1) * kStep);
    }
    if ((ic & 1) == 0 && (ic + 2) < NC) {
      __builtin_prefetch((const void*)(Arow + (ic + 2) * 32), 0, 3);
    }
    wmma4(c, a[pcur], b[pcur]);
  }

  // C/D layout: N = lane&15, M = r + 8*(lane>=16); fuse bias add
  const int rowb = mbase + wave * 16 + (laneHi ? 8 : 0);
#pragma unroll
  for (int j = 0; j < 4; ++j) {
    int col  = nbase + j * 16 + laneLo;
    float bi = bias[col];
#pragma unroll
    for (int r = 0; r < 8; ++r) {
      C[(size_t)(rowb + r) * kH + col] = c[j][r] + bi;
    }
  }
}

// --- LIF scan over T, emit spikes as bf16 (exact 0.0 / 1.0) ----------------
__global__ __launch_bounds__(256) void k_lif_spikes(const float* __restrict__ cur,
                                                    unsigned short* __restrict__ sp) {
  int tid = blockIdx.x * 256 + threadIdx.x;   // < 65536
  int b = tid >> 9, h = tid & 511;
  size_t base = (size_t)b * kH + h;
  float v = 0.f;
  for (int t = 0; t < kT; ++t) {
    size_t idx = (size_t)t * (kB * kH) + base;
    float I = cur[idx];
    v += (I - v) * 0.5f;                       // tau = 2
    bool s = (v >= 1.0f);
    sp[idx] = s ? (unsigned short)0x3F80 : (unsigned short)0;  // bf16 1.0 / 0.0
    v = s ? 0.f : v;                           // hard reset
  }
}

// --- LIF scan over T, accumulate spike count (time-sum of s2) --------------
__global__ __launch_bounds__(256) void k_lif_accum(const float* __restrict__ cur,
                                                   float* __restrict__ acc) {
  int tid = blockIdx.x * 256 + threadIdx.x;
  int b = tid >> 9, h = tid & 511;
  size_t base = (size_t)b * kH + h;
  float v = 0.f, a = 0.f;
  for (int t = 0; t < kT; ++t) {
    float I = cur[(size_t)t * (kB * kH) + base];
    v += (I - v) * 0.5f;
    if (v >= 1.0f) { a += 1.0f; v = 0.f; }
  }
  acc[base] = a;
}

// --- out[b][o] = acc[b,:]·W3[o,:] + T*b3[o] --------------------------------
__global__ __launch_bounds__(256) void k_out(const float* __restrict__ acc,
                                             const float* __restrict__ W3,
                                             const float* __restrict__ b3,
                                             float* __restrict__ out) {
  int tid = blockIdx.x * 256 + threadIdx.x;
  if (tid >= kB * kO) return;
  int b = tid / kO, o = tid % kO;
  const float* av = acc + (size_t)b * kH;
  const float* wv = W3 + (size_t)o * kH;
  float d = 0.f;
  for (int h = 0; h < kH; ++h) d += av[h] * wv[h];
  out[tid] = d + (float)kT * b3[o];
}

// --- workspace layout (bytes, 256-aligned) ---------------------------------
static constexpr size_t XB_OFF  = 0;                                  // 90,112,000
static constexpr size_t XB_B    = (size_t)kM * kDP * 2;
static constexpr size_t W1F_OFF = XB_OFF + XB_B;                      //    720,896
static constexpr size_t W1F_B   = (size_t)(kDP / 32) * kN16 * 1024;
static constexpr size_t W2F_OFF = W1F_OFF + W1F_B;                    //    524,288
static constexpr size_t W2F_B   = (size_t)(kH / 32) * kN16 * 1024;
static constexpr size_t CUR_OFF = W2F_OFF + W2F_B;                    // 131,072,000
static constexpr size_t CUR_B   = (size_t)kM * kH * 4;
static constexpr size_t ACC_OFF = CUR_OFF + CUR_B;                    //    262,144
// s1b (bf16, kM*kH*2 = 65.5 MB) aliases the dead xb region.

extern "C" void kernel_launch(void* const* d_in, const int* in_sizes, int n_in,
                              void* d_out, int out_size, void* d_ws, size_t ws_size,
                              hipStream_t stream) {
  (void)in_sizes; (void)n_in; (void)out_size; (void)ws_size;
  const float* x  = (const float*)d_in[0];
  const float* W1 = (const float*)d_in[1];
  const float* b1 = (const float*)d_in[2];
  const float* W2 = (const float*)d_in[3];
  const float* b2 = (const float*)d_in[4];
  const float* W3 = (const float*)d_in[5];
  const float* b3 = (const float*)d_in[6];
  float* out = (float*)d_out;

  char* ws = (char*)d_ws;
  unsigned short* xb  = (unsigned short*)(ws + XB_OFF);
  unsigned short* w1f = (unsigned short*)(ws + W1F_OFF);
  unsigned short* w2f = (unsigned short*)(ws + W2F_OFF);
  float*          cur = (float*)(ws + CUR_OFF);
  float*          acc = (float*)(ws + ACC_OFF);
  unsigned short* s1b = (unsigned short*)(ws + XB_OFF);  // alias: xb dead after GEMM1

  // 0) precision/layout conversion passes
  {
    unsigned int total = (unsigned int)kM * kDP;           // 45,056,000
    k_convert_x<<<(total + 255u) / 256u, 256, 0, stream>>>(x, xb);
  }
  k_wfrag<<<((kDP / 32) * kN16 * 512 + 255) / 256, 256, 0, stream>>>(
      W1, w1f, kDP / 32, kD, kD);
  k_wfrag<<<((kH / 32) * kN16 * 512 + 255) / 256, 256, 0, stream>>>(
      W2, w2f, kH / 32, kH, kH);

  dim3 gGemm(kM / 128, kH / 64);   // (500, 8)

  // 1) cur1 = xb * W1^T + b1
  k_gemm_bf16<kDP><<<gGemm, 256, 0, stream>>>(xb, w1f, b1, cur);

  // 2) LIF scan 1 -> spikes (bf16), aliased over xb
  k_lif_spikes<<<(kB * kH) / 256, 256, 0, stream>>>(cur, s1b);

  // 3) cur2 = s1 * W2^T + b2 (overwrites cur in place)
  k_gemm_bf16<kH><<<gGemm, 256, 0, stream>>>(s1b, w2f, b2, cur);

  // 4) LIF scan 2 -> time-summed spike counts [B,H]
  k_lif_accum<<<(kB * kH) / 256, 256, 0, stream>>>(cur, acc);

  // 5) out = acc * W3^T + T*b3
  k_out<<<(kB * kO + 255) / 256, 256, 0, stream>>>(acc, W3, b3, out);
}
